// ProteinNet_18219251270263
// MI455X (gfx1250) — matrix operand order repaired and checked by reading, not compiled, not run on hardware
//
#include <hip/hip_runtime.h>

// ProteinNet EdgeConv GNN for MI455X (gfx1250, wave32, WMMA).
// Heavy math = per-edge MLPs done as f16 WMMA GEMMs with f32 accumulation:
//   GEMM1: [E x 384] x [384 x 256]   (A gathered from x[dst] | x[src] | e)
//   GEMM2: [E x 256] x [256 x 128]   + BN/residual/scatter-add epilogue
// Roofline: 2.1e11 FLOPs, ~1.5 GB traffic -> ~70us mem @23.3TB/s, ~40us f16 WMMA.
// Weight chunks staged into LDS via async global->LDS copies; N-tiles processed in
// pairs (two independent WMMA chains) so LDS load latency overlaps the matrix pipe.

#define H       128
#define LN_EPS  1e-5f
#define BN_EPS  1e-5f
#define K1      384   // GEMM1 K
#define N1      256   // GEMM1 N
#define K2      256   // GEMM2 K
#define N2      128   // GEMM2 N

typedef __attribute__((ext_vector_type(16))) _Float16 v16h;
typedef __attribute__((ext_vector_type(8)))  _Float16 v8h;
typedef __attribute__((ext_vector_type(8)))  float    v8f;
typedef __attribute__((ext_vector_type(4)))  int      v4i;

#if defined(__gfx1250__) && __has_builtin(__builtin_amdgcn_global_load_async_to_lds_b128) && \
    __has_builtin(__builtin_amdgcn_s_wait_asynccnt)
#define HAVE_ASYNC_LDS 1
typedef __attribute__((address_space(1))) v4i g_v4i;   // global int4
typedef __attribute__((address_space(3))) v4i l_v4i;   // LDS int4
#else
#if defined(__gfx1250__)
#warning "gfx1250 async LDS builtins NOT available - using synchronous LDS staging"
#endif
#endif

// stage a [rows x 64]-halfs chunk (cols [c*64, c*64+64) of a row-major [rows x ncols]
// f16 matrix) into LDS; 16B per lane per iteration
template <int ROWS, int NCOLS, int NTHREADS>
__device__ __forceinline__ void stage_chunk(_Float16* sW, const _Float16* __restrict__ Wg,
                                            int c, int tid) {
#if defined(HAVE_ASYNC_LDS)
    for (int i = tid; i < (ROWS * 64) / 8; i += NTHREADS) {
        int row = i >> 3, cg = i & 7;
        __builtin_amdgcn_global_load_async_to_lds_b128(
            (g_v4i*)&Wg[row * NCOLS + c * 64 + cg * 8],
            (l_v4i*)&sW[row * 64 + cg * 8],
            0, 0);
    }
    __builtin_amdgcn_s_wait_asynccnt(0);
#else
    for (int i = tid; i < (ROWS * 64) / 8; i += NTHREADS) {
        int row = i >> 3, cg = i & 7;
        *(v8h*)&sW[row * 64 + cg * 8] = *(const v8h*)&Wg[row * NCOLS + c * 64 + cg * 8];
    }
#endif
}

// B fragment: 32x16 f16 tile; lane holds column N=lane&15, K = 8h+i / 16+8h+i (h=lane>>4)
__device__ __forceinline__ v16h load_bfrag(const _Float16* sW, int ks, int nt, int lane) {
    const _Float16* bp = &sW[(ks * 32 + (lane >> 4) * 8) * 64 + nt * 16 + (lane & 15)];
    v16h B;
    #pragma unroll
    for (int i = 0; i < 8; ++i) { B[i] = bp[i * 64]; B[i + 8] = bp[(16 + i) * 64]; }
    return B;
}

__device__ __forceinline__ v8f bias_acc(float b) {
    v8f a;
    #pragma unroll
    for (int i = 0; i < 8; ++i) a[i] = b;
    return a;
}

// ---------------------------------------------------------------- utilities

__global__ void f32_to_f16_k(const float* __restrict__ s, _Float16* __restrict__ d, int n) {
    int i = blockIdx.x * blockDim.x + threadIdx.x;
    if (i < n) d[i] = (_Float16)s[i];
}

__global__ void zero_f32_k(float* __restrict__ p, long n) {
    long i = (long)blockIdx.x * blockDim.x + threadIdx.x;
    if (i < n) p[i] = 0.f;
}

// ------------------------------------------------- node embed: emb->relu->lin->LN
__global__ __launch_bounds__(H)
void node_embed_k(const int* __restrict__ tok, const float* __restrict__ emb,
                  const float* __restrict__ W, const float* __restrict__ b,
                  const float* __restrict__ g, const float* __restrict__ beta,
                  float* __restrict__ x, _Float16* __restrict__ xh) {
    int n = blockIdx.x;
    int t = threadIdx.x;
    __shared__ float  s_in[H];
    __shared__ float2 s_r2[H];
    float v = emb[(long)tok[n] * H + t];
    s_in[t] = fmaxf(v, 0.f);
    __syncthreads();
    float acc = b[t];
    #pragma unroll 4
    for (int k = 0; k < H; ++k) acc += s_in[k] * W[k * H + t];
    s_r2[t] = make_float2(acc, acc * acc);
    __syncthreads();
    for (int s = H / 2; s > 0; s >>= 1) {
        if (t < s) { s_r2[t].x += s_r2[t + s].x; s_r2[t].y += s_r2[t + s].y; }
        __syncthreads();
    }
    float mean = s_r2[0].x / H;
    float var  = s_r2[0].y / H - mean * mean;
    float o = (acc - mean) * rsqrtf(var + LN_EPS) * g[t] + beta[t];
    x[(long)n * H + t]  = o;
    xh[(long)n * H + t] = (_Float16)o;
}

// ------------------------------------------------- edge embed: lin->relu->lin->LN
__global__ __launch_bounds__(H)
void edge_embed_k(const float* __restrict__ ea,
                  const float* __restrict__ W1, const float* __restrict__ b1,
                  const float* __restrict__ W2, const float* __restrict__ b2,
                  const float* __restrict__ g, const float* __restrict__ beta,
                  float* __restrict__ e, _Float16* __restrict__ eh) {
    long ed = blockIdx.x;
    int t = threadIdx.x;
    __shared__ float  s_in[H];
    __shared__ float2 s_r2[H];
    float a0 = ea[ed * 2 + 0], a1 = ea[ed * 2 + 1];
    s_in[t] = fmaxf(a0 * W1[t] + a1 * W1[H + t] + b1[t], 0.f);
    __syncthreads();
    float acc = b2[t];
    #pragma unroll 4
    for (int k = 0; k < H; ++k) acc += s_in[k] * W2[k * H + t];
    s_r2[t] = make_float2(acc, acc * acc);
    __syncthreads();
    for (int s = H / 2; s > 0; s >>= 1) {
        if (t < s) { s_r2[t].x += s_r2[t + s].x; s_r2[t].y += s_r2[t + s].y; }
        __syncthreads();
    }
    float mean = s_r2[0].x / H;
    float var  = s_r2[0].y / H - mean * mean;
    float o = (acc - mean) * rsqrtf(var + LN_EPS) * g[t] + beta[t];
    e[ed * H + t]  = o;
    eh[ed * H + t] = (_Float16)o;
}

// -------------------------------------------------------------- WMMA GEMM 1
// feature pointer for concat[x[dst] | x[src] | e[edge]], 8-half aligned chunks
__device__ __forceinline__ const _Float16*
feat_ptr(int k, const _Float16* xh, const _Float16* eh, int srcn, int dstn, long edge) {
    if (k < H)     return xh + (long)dstn * H + k;
    if (k < 2 * H) return xh + (long)srcn * H + (k - H);
    return eh + edge * H + (k - 2 * H);
}

__global__ __launch_bounds__(256)
void gemm1_k(const int* __restrict__ srcv, const int* __restrict__ dstv,
             const _Float16* __restrict__ xh, const _Float16* __restrict__ eh,
             const _Float16* __restrict__ W1h, const float* __restrict__ b1,
             _Float16* __restrict__ hbuf, int E) {
    __shared__ _Float16 sW[K1 * 64];                 // 48 KB weight chunk
    __shared__ float    sB[N1];                      // bias stage
    const int tid = threadIdx.x;
    const int lane = tid & 31, wave = tid >> 5;
    const long e0 = (long)blockIdx.x * 128 + wave * 16;
    const int m = lane & 15;
    long edge = e0 + m;
    long edge_c = edge < E ? edge : (long)(E - 1);
    const int srcn = srcv[edge_c], dstn = dstv[edge_c];
    const int khalf = (lane >> 4) * 8;               // ISA 16-bit A 16x32 lane layout
    const bool full = (e0 + 16 <= E);                // wave-uniform tail guard

    sB[tid] = b1[tid];                               // N1 == 256 == blockDim

    v16h A[12];
    #pragma unroll
    for (int ks = 0; ks < 12; ++ks) {
        int kb = ks * 32;
        v8h lo = *(const v8h*)feat_ptr(kb + khalf,      xh, eh, srcn, dstn, edge_c);
        v8h hi = *(const v8h*)feat_ptr(kb + 16 + khalf, xh, eh, srcn, dstn, edge_c);
        #pragma unroll
        for (int i = 0; i < 8; ++i) { A[ks][i] = lo[i]; A[ks][i + 8] = hi[i]; }
    }

    for (int c = 0; c < 4; ++c) {                    // 64-col weight chunks
        __syncthreads();
        stage_chunk<K1, N1, 256>(sW, W1h, c, tid);
        if (c + 1 < 4)                               // hint next chunk into caches
            __builtin_prefetch(&W1h[tid * N1 + (c + 1) * 64], 0, 1);
        __syncthreads();
        for (int ntp = 0; ntp < 2; ++ntp) {          // N-tile pairs: 2 indep WMMA chains
            const int nt0 = ntp * 2, nt1 = ntp * 2 + 1;
            const int n0a = c * 64 + nt0 * 16, n0b = c * 64 + nt1 * 16;
            v8f acc0 = bias_acc(sB[n0a + (lane & 15)]);
            v8f acc1 = bias_acc(sB[n0b + (lane & 15)]);
            v16h B0 = load_bfrag(sW, 0, nt0, lane);
            v16h B1 = load_bfrag(sW, 0, nt1, lane);
            #pragma unroll
            for (int ks = 0; ks < 11; ++ks) {
                v16h C0 = load_bfrag(sW, ks + 1, nt0, lane);
                v16h C1 = load_bfrag(sW, ks + 1, nt1, lane);
                acc0 = __builtin_amdgcn_wmma_f32_16x16x32_f16(
                        false, A[ks], false, B0, (short)0, acc0, false, false);
                acc1 = __builtin_amdgcn_wmma_f32_16x16x32_f16(
                        false, A[ks], false, B1, (short)0, acc1, false, false);
                B0 = C0; B1 = C1;
            }
            acc0 = __builtin_amdgcn_wmma_f32_16x16x32_f16(
                    false, A[11], false, B0, (short)0, acc0, false, false);
            acc1 = __builtin_amdgcn_wmma_f32_16x16x32_f16(
                    false, A[11], false, B1, (short)0, acc1, false, false);
            // C/D layout: M = r + 8*(lane>=16), N = lane&15; ReLU after linear1
            if (full) {
                #pragma unroll
                for (int r = 0; r < 8; ++r) {
                    long er = e0 + r + ((lane >> 4) << 3);
                    hbuf[er * N1 + n0a + (lane & 15)] = (_Float16)fmaxf(acc0[r], 0.f);
                    hbuf[er * N1 + n0b + (lane & 15)] = (_Float16)fmaxf(acc1[r], 0.f);
                }
            } else {
                #pragma unroll
                for (int r = 0; r < 8; ++r) {
                    long er = e0 + r + ((lane >> 4) << 3);
                    if (er < E) {
                        hbuf[er * N1 + n0a + (lane & 15)] = (_Float16)fmaxf(acc0[r], 0.f);
                        hbuf[er * N1 + n0b + (lane & 15)] = (_Float16)fmaxf(acc1[r], 0.f);
                    }
                }
            }
        }
    }
}

// -------------------------------------------------------------- WMMA GEMM 2
__global__ __launch_bounds__(256)
void gemm2_k(const int* __restrict__ dstv,
             const _Float16* __restrict__ hbuf,
             const _Float16* __restrict__ W2h, const float* __restrict__ b2,
             const float* __restrict__ bng, const float* __restrict__ bnb,
             const float* __restrict__ bnm, const float* __restrict__ bnv,
             float* __restrict__ xacc, float* __restrict__ e, _Float16* __restrict__ eh,
             int E, int do_relu) {
    __shared__ _Float16 sW[K2 * 64];                 // 32 KB weight chunk
    __shared__ float    sB[N2], sSc[N2], sSh[N2];    // bias + BN scale/shift stage
    const int tid = threadIdx.x;
    const int lane = tid & 31, wave = tid >> 5;
    const long e0 = (long)blockIdx.x * 128 + wave * 16;
    const int m = lane & 15;
    long edge = e0 + m;
    long edge_c = edge < E ? edge : (long)(E - 1);
    const int khalf = (lane >> 4) * 8;
    const bool full = (e0 + 16 <= E);

    if (tid < N2) {
        sB[tid] = b2[tid];
        float sc = bng[tid] * rsqrtf(bnv[tid] + BN_EPS);   // eval BatchNorm on e_out
        sSc[tid] = sc;
        sSh[tid] = bnb[tid] - bnm[tid] * sc;
    }

    const _Float16* arow = hbuf + edge_c * K2;
    v16h A[8];
    #pragma unroll
    for (int ks = 0; ks < 8; ++ks) {
        v8h lo = *(const v8h*)&arow[ks * 32 + khalf];
        v8h hi = *(const v8h*)&arow[ks * 32 + 16 + khalf];
        #pragma unroll
        for (int i = 0; i < 8; ++i) { A[ks][i] = lo[i]; A[ks][i + 8] = hi[i]; }
    }
    int dsts[8];
    #pragma unroll
    for (int r = 0; r < 8; ++r) {
        long er = e0 + r + ((lane >> 4) << 3);
        dsts[r] = dstv[er < E ? er : (long)(E - 1)];
    }

    for (int c = 0; c < 2; ++c) {
        __syncthreads();
        stage_chunk<K2, N2, 256>(sW, W2h, c, tid);
        __syncthreads();
        for (int ntp = 0; ntp < 2; ++ntp) {          // N-tile pairs: 2 indep WMMA chains
            const int nt0 = ntp * 2, nt1 = ntp * 2 + 1;
            const int na = c * 64 + nt0 * 16 + (lane & 15);
            const int nb = c * 64 + nt1 * 16 + (lane & 15);
            v8f acc0 = bias_acc(sB[na]);
            v8f acc1 = bias_acc(sB[nb]);
            v16h B0 = load_bfrag(sW, 0, nt0, lane);
            v16h B1 = load_bfrag(sW, 0, nt1, lane);
            #pragma unroll
            for (int ks = 0; ks < 7; ++ks) {
                v16h C0 = load_bfrag(sW, ks + 1, nt0, lane);
                v16h C1 = load_bfrag(sW, ks + 1, nt1, lane);
                acc0 = __builtin_amdgcn_wmma_f32_16x16x32_f16(
                        false, A[ks], false, B0, (short)0, acc0, false, false);
                acc1 = __builtin_amdgcn_wmma_f32_16x16x32_f16(
                        false, A[ks], false, B1, (short)0, acc1, false, false);
                B0 = C0; B1 = C1;
            }
            acc0 = __builtin_amdgcn_wmma_f32_16x16x32_f16(
                    false, A[7], false, B0, (short)0, acc0, false, false);
            acc1 = __builtin_amdgcn_wmma_f32_16x16x32_f16(
                    false, A[7], false, B1, (short)0, acc1, false, false);

            float sca = sSc[na], sha = sSh[na];
            float scb = sSc[nb], shb = sSh[nb];
            if (full) {
                #pragma unroll
                for (int r = 0; r < 8; ++r) {
                    long er = e0 + r + ((lane >> 4) << 3);
                    float va = acc0[r], vb = acc1[r];            // raw per-edge messages
                    atomicAdd(&xacc[(long)dsts[r] * H + na], va); // segment_sum (pre-BN)
                    atomicAdd(&xacc[(long)dsts[r] * H + nb], vb);
                    float ea2 = e[er * H + na] + (va * sca + sha); // e residual after BN
                    float eb2 = e[er * H + nb] + (vb * scb + shb);
                    if (do_relu) { ea2 = fmaxf(ea2, 0.f); eb2 = fmaxf(eb2, 0.f); }
                    e[er * H + na]  = ea2;  e[er * H + nb]  = eb2;
                    eh[er * H + na] = (_Float16)ea2;
                    eh[er * H + nb] = (_Float16)eb2;
                }
            } else {
                #pragma unroll
                for (int r = 0; r < 8; ++r) {
                    long er = e0 + r + ((lane >> 4) << 3);
                    if (er < E) {
                        float va = acc0[r], vb = acc1[r];
                        atomicAdd(&xacc[(long)dsts[r] * H + na], va);
                        atomicAdd(&xacc[(long)dsts[r] * H + nb], vb);
                        float ea2 = e[er * H + na] + (va * sca + sha);
                        float eb2 = e[er * H + nb] + (vb * scb + shb);
                        if (do_relu) { ea2 = fmaxf(ea2, 0.f); eb2 = fmaxf(eb2, 0.f); }
                        e[er * H + na]  = ea2;  e[er * H + nb]  = eb2;
                        eh[er * H + na] = (_Float16)ea2;
                        eh[er * H + nb] = (_Float16)eb2;
                    }
                }
            }
        }
    }
}

// ------------------------------------------------- node update: x += BN(sum); relu
__global__ void node_update_k(float* __restrict__ x, _Float16* __restrict__ xh,
                              const float* __restrict__ xacc,
                              const float* __restrict__ g, const float* __restrict__ b,
                              const float* __restrict__ mu, const float* __restrict__ vv,
                              long total, int do_relu) {
    long i = (long)blockIdx.x * blockDim.x + threadIdx.x;
    if (i >= total) return;
    int k = (int)(i & (H - 1));
    float sc = g[k] * rsqrtf(vv[k] + BN_EPS);
    float sh = b[k] - mu[k] * sc;
    float xn = x[i] + (xacc[i] * sc + sh);
    if (do_relu) xn = fmaxf(xn, 0.f);
    x[i]  = xn;
    xh[i] = (_Float16)xn;
}

// ------------------------------------------------------------- output head
__global__ __launch_bounds__(H)
void out_proj_k(const float* __restrict__ x, const float* __restrict__ W,
                const float* __restrict__ b, float* __restrict__ out) {
    int n = blockIdx.x, t = threadIdx.x;
    __shared__ float s[H];
    s[t] = x[(long)n * H + t];
    __syncthreads();
    if (t < 20) {
        float acc = b[t];
        #pragma unroll 4
        for (int k = 0; k < H; ++k) acc += s[k] * W[k * 20 + t];
        out[(long)n * 20 + t] = acc;
    }
}

// ---------------------------------------------------------------- launcher
extern "C" void kernel_launch(void* const* d_in, const int* in_sizes, int n_in,
                              void* d_out, int out_size, void* d_ws, size_t ws_size,
                              hipStream_t stream) {
    const int*   x_tokens   = (const int*)d_in[0];
    const int*   edge_index = (const int*)d_in[1];
    const float* edge_attr  = (const float*)d_in[2];
    const float* p_embed    = (const float*)d_in[3];
    const float* emb_W      = (const float*)d_in[4];
    const float* emb_b      = (const float*)d_in[5];
    const float* emb_g      = (const float*)d_in[6];
    const float* emb_beta   = (const float*)d_in[7];
    const float* adj_W1     = (const float*)d_in[8];
    const float* adj_b1     = (const float*)d_in[9];
    const float* adj_W2     = (const float*)d_in[10];
    const float* adj_b2     = (const float*)d_in[11];
    const float* adj_g      = (const float*)d_in[12];
    const float* adj_beta   = (const float*)d_in[13];
    const float* out_W      = (const float*)d_in[n_in - 2];
    const float* out_b      = (const float*)d_in[n_in - 1];

    const int N = in_sizes[0];
    const int E = in_sizes[1] / 2;
    const int* srcv = edge_index;
    const int* dstv = edge_index + E;

    // workspace carve-out (~270 MB)
    char* ws = (char*)d_ws;
    size_t off = 0;
    auto alloc = [&](size_t bytes) -> void* {
        void* p = (void*)(ws + off);
        off = (off + bytes + 255) & ~(size_t)255;
        return p;
    };
    _Float16* xh   = (_Float16*)alloc((size_t)N * H * 2);
    float*    x    = (float*)   alloc((size_t)N * H * 4);
    float*    xacc = (float*)   alloc((size_t)N * H * 4);
    float*    e    = (float*)   alloc((size_t)E * H * 4);
    _Float16* eh   = (_Float16*)alloc((size_t)E * H * 2);
    _Float16* hbuf = (_Float16*)alloc((size_t)E * N1 * 2);
    _Float16* W1h[4];
    _Float16* W2h[4];
    for (int l = 0; l < 4; ++l) {
        W1h[l] = (_Float16*)alloc((size_t)K1 * N1 * 2);
        W2h[l] = (_Float16*)alloc((size_t)K2 * N2 * 2);
    }
    (void)ws_size; (void)out_size;

    // per-layer param table: base = 14 + 12*l
    auto cp = [&](int l, int j) -> const float* { return (const float*)d_in[14 + 12 * l + j]; };

    // 1) weights -> f16
    for (int l = 0; l < 4; ++l) {
        f32_to_f16_k<<<(K1 * N1 + 255) / 256, 256, 0, stream>>>(cp(l, 0), W1h[l], K1 * N1);
        f32_to_f16_k<<<(K2 * N2 + 255) / 256, 256, 0, stream>>>(cp(l, 2), W2h[l], K2 * N2);
    }
    // 2) node + edge embeddings
    node_embed_k<<<N, H, 0, stream>>>(x_tokens, p_embed, emb_W, emb_b, emb_g, emb_beta, x, xh);
    edge_embed_k<<<E, H, 0, stream>>>(edge_attr, adj_W1, adj_b1, adj_W2, adj_b2,
                                      adj_g, adj_beta, e, eh);
    // 3) four EdgeConv layers
    const int gemm_blocks = (E + 127) / 128;
    for (int l = 0; l < 4; ++l) {
        int do_relu = (l < 3) ? 1 : 0;
        zero_f32_k<<<((long)N * H + 255) / 256, 256, 0, stream>>>(xacc, (long)N * H);
        gemm1_k<<<gemm_blocks, 256, 0, stream>>>(srcv, dstv, xh, eh, W1h[l], cp(l, 1), hbuf, E);
        gemm2_k<<<gemm_blocks, 256, 0, stream>>>(dstv, hbuf, W2h[l], cp(l, 3),
                                                 cp(l, 8), cp(l, 9), cp(l, 10), cp(l, 11),
                                                 xacc, e, eh, E, do_relu);
        node_update_k<<<((long)N * H + 255) / 256, 256, 0, stream>>>(
            x, xh, xacc, cp(l, 4), cp(l, 5), cp(l, 6), cp(l, 7), (long)N * H, do_relu);
    }
    // 4) output projection
    out_proj_k<<<N, H, 0, stream>>>(x, out_W, out_b, (float*)d_out);
}
